// FAISSIndexWrapper_14010183320249
// MI455X (gfx1250) — compile-verified, matrix operand order checked by reading, not run on hardware
//
#include <hip/hip_runtime.h>
#include <hip/hip_bf16.h>

// Problem constants (from reference): B=512, N=262144, D=512, K=10
#define BQ   512
#define NDB  262144
#define DIM  512
#define TOPK 10

#define ROW_TILE   32          // rows of image_embs per block (2 M-tiles per wave)
#define COL_STEP   128         // columns of index per inner step (8 waves * 16)
#define COL_BLK    4096        // columns per block (outer)
#define NUM_CB     (NDB / COL_BLK)      // 64 column blocks
#define NUM_RT     (BQ / ROW_TILE)      // 16 row tiles
#define STEPS      (COL_BLK / COL_STEP) // 32 inner steps
#define KSTEPS     (DIM / 32)           // 16 wmma K-steps

typedef __attribute__((ext_vector_type(16))) __bf16 v16bf;
typedef __attribute__((ext_vector_type(8)))  __bf16 bf16x8;
typedef __attribute__((ext_vector_type(8)))  float  v8f;

struct Cand { float v; int i; };

union FragU { v16bf v; bf16x8 h[2]; };

__device__ inline bf16x8 cvt8(float4 a, float4 b) {
    bf16x8 r;
    r[0] = (__bf16)a.x; r[1] = (__bf16)a.y; r[2] = (__bf16)a.z; r[3] = (__bf16)a.w;
    r[4] = (__bf16)b.x; r[5] = (__bf16)b.y; r[6] = (__bf16)b.z; r[7] = (__bf16)b.w;
    return r;
}

// Fully-unrolled, register-resident top-10 insert. Strictly-greater compares
// mean the earlier (lower-index) candidate wins ties, matching jax.lax.top_k.
__device__ inline void topk_insert(float v, int idx, float (&tv)[TOPK], int (&ti)[TOPK]) {
    if (v > tv[TOPK - 1]) {
        tv[TOPK - 1] = v; ti[TOPK - 1] = idx;
#pragma unroll
        for (int q = TOPK - 1; q >= 1; --q) {
            if (tv[q] > tv[q - 1]) {
                float fv = tv[q]; tv[q] = tv[q - 1]; tv[q - 1] = fv;
                int   fi = ti[q]; ti[q] = ti[q - 1]; ti[q - 1] = fi;
            }
        }
    }
}

__global__ __launch_bounds__(256)
void gemm_topk_kernel(const float* __restrict__ imageE,
                      const float* __restrict__ indexM,
                      Cand* __restrict__ cands) {
    __shared__ alignas(16) __bf16 Abf[ROW_TILE][DIM];       // 32 KB
    __shared__ alignas(16) float  sims[ROW_TILE][COL_STEP]; // 16 KB

    const int tid  = threadIdx.x;
    const int wave = tid >> 5;        // 0..7
    const int lane = tid & 31;        // 0..31
    const int rm   = blockIdx.x * ROW_TILE;
    const int cb0  = blockIdx.y * COL_BLK;

    // ---- Stage A tile (32 x 512) into LDS as bf16 ----
    {
        const int total = ROW_TILE * DIM;          // 16384 elements
        for (int e = tid; e < total; e += 256) {
            int r = e >> 9;          // e / DIM
            int k = e & (DIM - 1);
            Abf[r][k] = (__bf16)imageE[(size_t)(rm + r) * DIM + k];
        }
    }

    // Per-thread running top-K (row t owned by thread t, t < 32 == wave 0)
    float tv[TOPK];
    int   ti[TOPK];
#pragma unroll
    for (int q = 0; q < TOPK; ++q) { tv[q] = -3.4e38f; ti[q] = 0; }

    __syncthreads();

    const int mrow  = lane & 15;          // M (A) / N (B) sub-index
    const int khalf = (lane >> 4) * 8;    // K0 per 16-bit fragment layout
    const __bf16* arow0 = &Abf[mrow][0];        // M-tile 0: rows 0..15
    const __bf16* arow1 = &Abf[16 + mrow][0];   // M-tile 1: rows 16..31

    for (int s = 0; s < STEPS; ++s) {
        const int colBase = cb0 + s * COL_STEP;
        const int nCol    = colBase + wave * 16 + mrow;   // database row for B frag
        const float* brow = indexM + (size_t)nCol * DIM;

        v8f acc0 = {};
        v8f acc1 = {};

#pragma unroll 4
        for (int kk = 0; kk < KSTEPS; ++kk) {
            const int k = kk * 32;

            // A fragments from LDS: two ds_load_b128 per tile per lane
            FragU a0, a1;
            a0.h[0] = *(const bf16x8*)(arow0 + k + khalf);
            a0.h[1] = *(const bf16x8*)(arow0 + k + khalf + 16);
            a1.h[0] = *(const bf16x8*)(arow1 + k + khalf);
            a1.h[1] = *(const bf16x8*)(arow1 + k + khalf + 16);

            // B fragment from global f32, converted to bf16 on the fly;
            // shared by BOTH accumulator chains (halves cvt per FLOP).
            float4 f0 = *(const float4*)(brow + k + khalf);
            float4 f1 = *(const float4*)(brow + k + khalf + 4);
            float4 f2 = *(const float4*)(brow + k + khalf + 16);
            float4 f3 = *(const float4*)(brow + k + khalf + 20);

            if (k + 32 < DIM) {
                __builtin_prefetch(brow + k + 32 + khalf, 0, 3);
            }

            FragU b;
            b.h[0] = cvt8(f0, f1);
            b.h[1] = cvt8(f2, f3);

            // Two independent WMMA chains -> no WMMA->WMMA RAW hazard NOPs
            acc0 = __builtin_amdgcn_wmma_f32_16x16x32_bf16(
                false, a0.v, false, b.v, (short)0, acc0, false, false);
            acc1 = __builtin_amdgcn_wmma_f32_16x16x32_bf16(
                false, a1.v, false, b.v, (short)0, acc1, false, false);
        }

        // Wait for previous step's top-k scan to finish reading sims
        __syncthreads();

        // Spill C tiles: VGPR r, lane L -> row = r + 8*(L/16), col = L%16
        {
            const int chi = (lane >> 4) * 8;
#pragma unroll
            for (int r = 0; r < 8; ++r) {
                sims[r + chi][wave * 16 + mrow]      = acc0[r];
                sims[16 + r + chi][wave * 16 + mrow] = acc1[r];
            }
        }
        __syncthreads();

        // Threads 0..31 (wave 0): scan 128 columns for their row
        if (tid < ROW_TILE) {
            for (int j = 0; j < COL_STEP; ++j) {
                topk_insert(sims[tid][j], colBase + j, tv, ti);
            }
        }
    }

    __syncthreads();

    if (tid < ROW_TILE) {
        Cand* out = cands + ((size_t)(blockIdx.x * NUM_CB + blockIdx.y) * ROW_TILE + tid) * TOPK;
#pragma unroll
        for (int q = 0; q < TOPK; ++q) {
            out[q].v = tv[q];
            out[q].i = ti[q];
        }
    }
}

__global__ __launch_bounds__(32)
void merge_topk_kernel(const Cand* __restrict__ cands, int* __restrict__ out) {
    const int row = blockIdx.x;
    if (threadIdx.x != 0) return;

    const int rowTile = row / ROW_TILE;
    const int r       = row % ROW_TILE;

    float tv[TOPK];
    int   ti[TOPK];
#pragma unroll
    for (int q = 0; q < TOPK; ++q) { tv[q] = -3.4e38f; ti[q] = 0; }

    for (int cb = 0; cb < NUM_CB; ++cb) {
        const Cand* c = cands + ((size_t)(rowTile * NUM_CB + cb) * ROW_TILE + r) * TOPK;
        for (int j = 0; j < TOPK; ++j) {
            topk_insert(c[j].v, c[j].i, tv, ti);
        }
    }

#pragma unroll
    for (int q = 0; q < TOPK; ++q) {
        out[row * TOPK + q] = ti[q];
    }
}

extern "C" void kernel_launch(void* const* d_in, const int* in_sizes, int n_in,
                              void* d_out, int out_size, void* d_ws, size_t ws_size,
                              hipStream_t stream) {
    (void)in_sizes; (void)n_in; (void)out_size; (void)ws_size;

    const float* imageE = (const float*)d_in[0];   // [512, 512] f32
    const float* indexM = (const float*)d_in[1];   // [262144, 512] f32
    // d_in[2] is k (==10), a compile-time constant here.

    int*  out   = (int*)d_out;                     // [512, 10] int32
    Cand* cands = (Cand*)d_ws;                     // 16*64*32*10*8 B = 2.62 MB

    dim3 grid(NUM_RT, NUM_CB);                     // x fastest: all row-tiles of a
                                                   // column chunk run adjacently -> L2 reuse
    gemm_topk_kernel<<<grid, 256, 0, stream>>>(imageE, indexM, cands);
    merge_topk_kernel<<<BQ, 32, 0, stream>>>(cands, out);
}
